// RBFLayer_5085241278855
// MI455X (gfx1250) — matrix operand order repaired and checked by reading, not compiled, not run on hardware
//
#include <hip/hip_runtime.h>
#include <math.h>

// ---- CDNA5 vector types ----
typedef __attribute__((ext_vector_type(16))) __bf16        v16bf;
typedef __attribute__((ext_vector_type(8)))  float         v8f;
typedef __attribute__((ext_vector_type(4)))  float         f32x4;
typedef __attribute__((ext_vector_type(4)))  int           i32x4;
typedef __attribute__((ext_vector_type(4)))  unsigned int  u32x4;
typedef __attribute__((ext_vector_type(2)))  unsigned int  u32x2;

#define N_ROWS 16384
#define M_COLS 2048
#define K_DIM  512

#define BN 128           // x rows per workgroup tile
#define BM 128           // center cols per workgroup tile
#define BK 64            // K chunk staged in LDS
#define NCHUNK (K_DIM / BK)
#define LDSK 72          // padded LDS row stride (elements), 144B = 16B-aligned

#if defined(__AMDGCN__) && __has_builtin(__builtin_amdgcn_global_load_async_to_lds_b128) && \
    __has_builtin(__builtin_amdgcn_s_wait_asynccnt)
#define HAS_ASYNC_LDS 1
#else
#define HAS_ASYNC_LDS 0
#endif

// fp32 -> bf16 (round to nearest even), and exact widening back
__device__ __forceinline__ unsigned short f2bf(float f) {
    unsigned int u = __float_as_uint(f);
    u += 0x7fffu + ((u >> 16) & 1u);
    return (unsigned short)(u >> 16);
}
__device__ __forceinline__ float bf2f(unsigned short h) {
    return __uint_as_float(((unsigned int)h) << 16);
}

union Frag {            // one WMMA 16-bit A/B operand: 8 VGPRs per lane
    v16bf v;
    u32x4 q[2];
};

// 16-byte global->LDS copy: async on CDNA5, sync fallback otherwise
__device__ __forceinline__ void cp16(const unsigned short* g, unsigned short* l) {
#if HAS_ASYNC_LDS
    __builtin_amdgcn_global_load_async_to_lds_b128(
        (__attribute__((address_space(1))) i32x4*)(g),
        (__attribute__((address_space(3))) i32x4*)(l), 0, 0);
#else
    *(u32x4*)l = *(const u32x4*)g;
#endif
}

// ---------- prologue: split fp32 -> bf16 hi/lo arrays ----------
__global__ __launch_bounds__(256) void convert_kernel(const float* __restrict__ src,
                                                      unsigned short* __restrict__ hi,
                                                      unsigned short* __restrict__ lo,
                                                      int n4) {
    int idx = blockIdx.x * 256 + threadIdx.x;
    if (idx >= n4) return;
    f32x4 v = *(const f32x4*)(src + (size_t)idx * 4);
    unsigned short h0 = f2bf(v[0]), h1 = f2bf(v[1]), h2 = f2bf(v[2]), h3 = f2bf(v[3]);
    unsigned short l0 = f2bf(v[0] - bf2f(h0)), l1 = f2bf(v[1] - bf2f(h1));
    unsigned short l2 = f2bf(v[2] - bf2f(h2)), l3 = f2bf(v[3] - bf2f(h3));
    u32x2 hp; hp[0] = (unsigned)h0 | ((unsigned)h1 << 16); hp[1] = (unsigned)h2 | ((unsigned)h3 << 16);
    u32x2 lp; lp[0] = (unsigned)l0 | ((unsigned)l1 << 16); lp[1] = (unsigned)l2 | ((unsigned)l3 << 16);
    *(u32x2*)(hi + (size_t)idx * 4) = hp;
    *(u32x2*)(lo + (size_t)idx * 4) = lp;
}

// ---------- prologue: row-wise sum of squares (one wave32 per row) ----------
__global__ __launch_bounds__(256) void rowsq_kernel(const float* __restrict__ src,
                                                    float* __restrict__ dst, int rows) {
    int row  = blockIdx.x * 8 + (threadIdx.x >> 5);
    int lane = threadIdx.x & 31;
    if (row >= rows) return;
    const float* p = src + (size_t)row * K_DIM;
    float s = 0.f;
#pragma unroll
    for (int i = 0; i < 4; ++i) {
        f32x4 v = *(const f32x4*)(p + lane * 4 + i * 128);
        s += v[0]*v[0] + v[1]*v[1] + v[2]*v[2] + v[3]*v[3];
    }
#pragma unroll
    for (int off = 16; off >= 1; off >>= 1) s += __shfl_xor(s, off, 32);
    if (lane == 0) dst[row] = s;
}

// ---------- prologue: 1/(2*sigma^2) = 0.5*exp(-2*log_sigma) ----------
__global__ __launch_bounds__(256) void sigma_kernel(const float* __restrict__ ls,
                                                    float* __restrict__ inv2s2, int m) {
    int t = blockIdx.x * blockDim.x + threadIdx.x;
    if (t < m) inv2s2[t] = 0.5f * __expf(-2.0f * ls[t]);
}

// ---------- main: split-bf16 WMMA GEMM, async double-buffered LDS, fused RBF epilogue ----------
// launch_bounds(256, 4): cap at 256 VGPRs/wave so 2 workgroups (4 waves/SIMD) co-reside per WGP
__global__ __launch_bounds__(256, 4) void rbf_wmma_kernel(
    const unsigned short* __restrict__ Xhi, const unsigned short* __restrict__ Xlo,
    const unsigned short* __restrict__ Chi, const unsigned short* __restrict__ Clo,
    const float* __restrict__ xsq,    // [N]
    const float* __restrict__ csq,    // [M]
    const float* __restrict__ inv2s2, // [M]
    float* __restrict__ out)          // [N, M]
{
    __shared__ __align__(16) unsigned short Ahi[2][BN][LDSK];
    __shared__ __align__(16) unsigned short Alo[2][BN][LDSK];
    __shared__ __align__(16) unsigned short Bhi[2][BM][LDSK];
    __shared__ __align__(16) unsigned short Blo[2][BM][LDSK];
    __shared__ float s_xsq[BN];
    __shared__ float s_csq[BM];
    __shared__ float s_inv[BM];

    const int t    = threadIdx.x;
    const int lane = t & 31;
    const int wave = t >> 5;

    const int rowBlock = blockIdx.y * BN;   // into x rows
    const int colBlock = blockIdx.x * BM;   // into center rows (output cols)

    if (t < BN) {
        s_xsq[t] = xsq[rowBlock + t];
    } else {
        int c = t - BN;
        s_csq[c] = csq[colBlock + c];
        s_inv[c] = inv2s2[colBlock + c];
    }

    const int wRow = (wave >> 2) * 64;   // 0 or 64
    const int wCol = (wave & 3)  * 32;   // 0,32,64,96

    v8f acc[4][2];
#pragma unroll
    for (int r = 0; r < 4; ++r)
#pragma unroll
        for (int c = 0; c < 2; ++c)
#pragma unroll
            for (int v = 0; v < 8; ++v) acc[r][c][v] = 0.0f;

    // ---- staging: 16 x 16B ops per thread per chunk (4 arrays x 4 pieces) ----
    auto stage = [&](int buf, int k0) {
#pragma unroll
        for (int j = 0; j < 4; ++j) {
            int idx = t + j * 256;          // 0..1023
            int row = idx >> 3;             // 0..127
            int ke  = (idx & 7) * 8;        // 0..56 step 8 (elements)
            size_t gx = (size_t)(rowBlock + row) * K_DIM + (k0 + ke);
            size_t gc = (size_t)(colBlock + row) * K_DIM + (k0 + ke);
            cp16(Xhi + gx, &Ahi[buf][row][ke]);
            cp16(Xlo + gx, &Alo[buf][row][ke]);
            cp16(Chi + gc, &Bhi[buf][row][ke]);
            cp16(Clo + gc, &Blo[buf][row][ke]);
        }
    };

    stage(0, 0);

    // unroll exactly by the double-buffer period: buf parity is compile-time per body,
    // code size and register pressure stay bounded (no VGPR-MSB paging)
#pragma unroll 2
    for (int ci = 0; ci < NCHUNK; ++ci) {
        const int buf = ci & 1;
        if (ci + 1 < NCHUNK) {
            stage(buf ^ 1, (ci + 1) * BK);
#if HAS_ASYNC_LDS
            __builtin_amdgcn_s_wait_asynccnt(16);  // retire current chunk; next stays in flight
#endif
        } else {
#if HAS_ASYNC_LDS
            __builtin_amdgcn_s_wait_asynccnt(0);
#endif
        }
        __syncthreads();

#pragma unroll
        for (int kk = 0; kk < BK; kk += 32) {
            // A 16-bit layout: lanes<16 hold K {kk..kk+7, kk+16..kk+23}; lanes>=16 offset +8
            const int kA = kk + ((lane & 16) ? 8 : 0);
            const int rA = wRow + (lane & 15);
            Frag ah[4], al[4];
#pragma unroll
            for (int r = 0; r < 4; ++r) {
                const unsigned short* ph = &Ahi[buf][rA + r * 16][kA];
                ah[r].q[0] = *(const u32x4*)(ph);
                ah[r].q[1] = *(const u32x4*)(ph + 16);
                const unsigned short* pl = &Alo[buf][rA + r * 16][kA];
                al[r].q[0] = *(const u32x4*)(pl);
                al[r].q[1] = *(const u32x4*)(pl + 16);
            }
            // B 16-bit layout: lanes<16 hold K {kk..kk+7, kk+16..kk+23}; lanes>=16 offset +8
            const int kB = kk + ((lane & 16) ? 8 : 0);
            const int cB = wCol + (lane & 15);
            Frag bh[2], bl[2];
#pragma unroll
            for (int c = 0; c < 2; ++c) {
                const unsigned short* ph = &Bhi[buf][cB + c * 16][kB];
                bh[c].q[0] = *(const u32x4*)(ph);
                bh[c].q[1] = *(const u32x4*)(ph + 16);
                const unsigned short* pl = &Blo[buf][cB + c * 16][kB];
                bl[c].q[0] = *(const u32x4*)(pl);
                bl[c].q[1] = *(const u32x4*)(pl + 16);
            }
            // split-bf16: acc += hi*hi + hi*lo + lo*hi   (lo*lo dropped, ~2^-18)
#pragma unroll
            for (int r = 0; r < 4; ++r) {
#pragma unroll
                for (int c = 0; c < 2; ++c) {
                    acc[r][c] = __builtin_amdgcn_wmma_f32_16x16x32_bf16(
                        false, ah[r].v, false, bh[c].v, (short)0, acc[r][c], false, false);
                    acc[r][c] = __builtin_amdgcn_wmma_f32_16x16x32_bf16(
                        false, ah[r].v, false, bl[c].v, (short)0, acc[r][c], false, false);
                    acc[r][c] = __builtin_amdgcn_wmma_f32_16x16x32_bf16(
                        false, al[r].v, false, bh[c].v, (short)0, acc[r][c], false, false);
                }
            }
        }
        __syncthreads();
    }

    // ---- fused RBF epilogue ----
    // C/D f32 16x16 layout: lane -> N column (lane&15), VGPR v -> M row v (+8 for lanes>=16)
    const int jc   = lane & 15;
    const int iOff = (lane & 16) ? 8 : 0;
#pragma unroll
    for (int r = 0; r < 4; ++r) {
#pragma unroll
        for (int c = 0; c < 2; ++c) {
            const int mLocal = wCol + c * 16 + jc;
            const float cq = s_csq[mLocal];
            const float iv = s_inv[mLocal];
            const size_t m = (size_t)(colBlock + mLocal);
#pragma unroll
            for (int v = 0; v < 8; ++v) {
                const int nLocal = wRow + r * 16 + iOff + v;
                const float d = s_xsq[nLocal] + cq - 2.0f * acc[r][c][v];
                // non-temporal: don't let the 128MB output evict reused tiles from L2
                __builtin_nontemporal_store(__expf(-d * iv),
                                            &out[(size_t)(rowBlock + nLocal) * M_COLS + m]);
            }
        }
    }
}

extern "C" void kernel_launch(void* const* d_in, const int* in_sizes, int n_in,
                              void* d_out, int out_size, void* d_ws, size_t ws_size,
                              hipStream_t stream) {
    const float* x       = (const float*)d_in[0];   // [16384, 512]
    const float* centers = (const float*)d_in[1];   // [2048, 512]
    const float* ls      = (const float*)d_in[2];   // [2048]
    float* out = (float*)d_out;                     // [16384, 2048]

    // workspace layout: xhi | xlo | chi | clo | xsq | csq | inv2s2  (~36 MB)
    unsigned short* xhi = (unsigned short*)d_ws;
    unsigned short* xlo = xhi + (size_t)N_ROWS * K_DIM;
    unsigned short* chi = xlo + (size_t)N_ROWS * K_DIM;
    unsigned short* clo = chi + (size_t)M_COLS * K_DIM;
    float* xsq = (float*)(clo + (size_t)M_COLS * K_DIM);
    float* csq = xsq + N_ROWS;
    float* inv = csq + M_COLS;

    const int xn4 = N_ROWS * K_DIM / 4;
    const int cn4 = M_COLS * K_DIM / 4;
    convert_kernel<<<xn4 / 256, 256, 0, stream>>>(x, xhi, xlo, xn4);
    convert_kernel<<<cn4 / 256, 256, 0, stream>>>(centers, chi, clo, cn4);
    rowsq_kernel<<<N_ROWS / 8, 256, 0, stream>>>(x, xsq, N_ROWS);
    rowsq_kernel<<<M_COLS / 8, 256, 0, stream>>>(centers, csq, M_COLS);
    sigma_kernel<<<(M_COLS + 255) / 256, 256, 0, stream>>>(ls, inv, M_COLS);

    dim3 grid(M_COLS / BM, N_ROWS / BN);  // (16, 128)
    rbf_wmma_kernel<<<grid, 256, 0, stream>>>(xhi, xlo, chi, clo, xsq, csq, inv, out);
}
